// MultiheadAttention_11605001634093
// MI455X (gfx1250) — matrix operand order repaired and checked by reading, not compiled, not run on hardware
//
#include <hip/hip_runtime.h>
#include <hip/hip_bf16.h>

// ---------------------------------------------------------------------------
// MultiheadAttention forward for MI455X (gfx1250, wave32, WMMA bf16).
// B=2, S=2048, D=1024, H=16, DK=64.
// ---------------------------------------------------------------------------

#define B_SZ   2
#define SEQ    2048
#define DMODEL 1024
#define NHEADS 16
#define DK     64
#define MTOT   (B_SZ * SEQ)          // 4096
#define NEGVAL (-1000000000.0f)

typedef __attribute__((ext_vector_type(16))) __bf16 v16bf;
typedef __attribute__((ext_vector_type(8)))  __bf16 v8bf;
typedef __attribute__((ext_vector_type(8)))  float  v8f;

// Async global->LDS copy support (gfx1250 GLOBAL_LOAD_ASYNC_TO_LDS_B128,
// tracked with ASYNCcnt). Parameter types per compiler diagnostic:
//   (int4-gccvec __device__ *src, int4-gccvec __shared__ *dst, imm, imm)
#if defined(__has_builtin)
#if __has_builtin(__builtin_amdgcn_global_load_async_to_lds_b128) && \
    __has_builtin(__builtin_amdgcn_s_wait_asynccnt)
#define HAVE_ASYNC_LDS 1
#endif
#endif

#ifdef HAVE_ASYNC_LDS
typedef int v4i_gcc __attribute__((vector_size(16)));
typedef __attribute__((address_space(1))) v4i_gcc* g_v4i_ptr;
typedef __attribute__((address_space(3))) v4i_gcc* l_v4i_ptr;
#define ASYNC_CP_B128(gsrc, ldst)                                         \
    __builtin_amdgcn_global_load_async_to_lds_b128(                       \
        (g_v4i_ptr)(gsrc), (l_v4i_ptr)(ldst), 0, 0)
#endif

__device__ __forceinline__ unsigned short f32_to_bf16_bits(float f) {
    unsigned int u = __float_as_uint(f);
    unsigned int r = u + 0x7FFFu + ((u >> 16) & 1u);   // round-to-nearest-even
    return (unsigned short)(r >> 16);
}

// ---- WMMA fragment loaders (per CDNA5 ISA 7.12.2 16-bit layouts) ----------
// A (16x32, MxK): lanes 0-15: M=lane, K={0..7,16..23}; lanes 16-31: M=lane-16,
// K={8..15,24..31}.
__device__ __forceinline__ v16bf load_fragA(const unsigned short* tile, int ld, int lane) {
    int m  = lane & 15;
    int kb = (lane & 16) ? 8 : 0;
    const unsigned short* p = tile + m * ld;
    v8bf lo = *(const v8bf*)(p + kb);
    v8bf hi = *(const v8bf*)(p + kb + 16);
    v16bf a;
#pragma unroll
    for (int i = 0; i < 8; ++i) { a[i] = lo[i]; a[i + 8] = hi[i]; }
    return a;
}
// B (32x16, KxN) stored in LDS as [n][k]: lanes 0-15: N=lane, K={0..7,16..23};
// lanes 16-31: N=lane-16, K={8..15,24..31}.
__device__ __forceinline__ v16bf load_fragB(const unsigned short* tile, int ld, int lane) {
    int n  = lane & 15;
    int kb = (lane & 16) ? 8 : 0;
    const unsigned short* p = tile + n * ld + kb;
    v8bf lo = *(const v8bf*)(p);
    v8bf hi = *(const v8bf*)(p + 16);
    v16bf b;
#pragma unroll
    for (int i = 0; i < 8; ++i) { b[i] = lo[i]; b[i + 8] = hi[i]; }
    return b;
}

// ---------------------------------------------------------------------------
// Kernel: fp32 -> bf16 bulk convert
// ---------------------------------------------------------------------------
__global__ void mha_cvt_bf16(const float* __restrict__ in,
                             unsigned short* __restrict__ out, int n) {
    int i = blockIdx.x * blockDim.x + threadIdx.x;
    if (i < n) out[i] = f32_to_bf16_bits(in[i]);
}

// ---------------------------------------------------------------------------
// Kernel: GEMM  Y[m,n] = sum_k A[m,k] * W[n,k] + bias[n]
// A: [M,K] bf16, W: [N,K] bf16.  Tiles 128x128x32, 8 waves, each wave
// 64x32 = 4x2 WMMA tiles.
// mode 0: write bf16 per-head layout out_h[((b*H+h)*S+s)*DK+d]
// mode 1: write fp32 out[m*N+n] = acc + bias[n] + residual[m*N+n]
// ---------------------------------------------------------------------------
#define LDT 40   // LDS row stride in u16 (32 data + 8 pad)

__global__ void mha_gemm_bf16(const unsigned short* __restrict__ A,
                              const unsigned short* __restrict__ W,
                              const float* __restrict__ bias,
                              const float* __restrict__ residual,
                              float* __restrict__ out_f32,
                              unsigned short* __restrict__ out_h,
                              int M, int N, int K, int mode) {
    __shared__ unsigned short As[128 * LDT];
    __shared__ unsigned short Bs[128 * LDT];

    const int t    = threadIdx.x;
    const int lane = t & 31;
    const int wave = t >> 5;            // 0..7
    const int wm   = wave >> 2;         // 0..1 -> 64 rows each
    const int wn   = wave & 3;          // 0..3 -> 32 cols each
    const int mBase = blockIdx.x * 128;
    const int nBase = blockIdx.y * 128;

    v8f acc[4][2];
#pragma unroll
    for (int i = 0; i < 4; ++i)
#pragma unroll
        for (int j = 0; j < 2; ++j) acc[i][j] = (v8f){0,0,0,0,0,0,0,0};

    const int lrow = t >> 1;            // 0..127
    const int lks  = (t & 1) * 16;      // 0 or 16

    for (int k0 = 0; k0 < K; k0 += 32) {
        const unsigned short* sa = A + (size_t)(mBase + lrow) * K + k0 + lks;
        const unsigned short* sb = W + (size_t)(nBase + lrow) * K + k0 + lks;
        __syncthreads();
#ifdef HAVE_ASYNC_LDS
        ASYNC_CP_B128(sa,     &As[lrow * LDT + lks]);
        ASYNC_CP_B128(sa + 8, &As[lrow * LDT + lks + 8]);
        ASYNC_CP_B128(sb,     &Bs[lrow * LDT + lks]);
        ASYNC_CP_B128(sb + 8, &Bs[lrow * LDT + lks + 8]);
        if (k0 + 32 < K) {
            __builtin_prefetch(sa + 32, 0, 1);
            __builtin_prefetch(sb + 32, 0, 1);
        }
        __builtin_amdgcn_s_wait_asynccnt(0);
#else
        {   // stage A/B tiles: 2 threads per row, 16 bf16 (=2x uint4) each
            uint4 a0 = *(const uint4*)(sa);
            uint4 a1 = *(const uint4*)(sa + 8);
            *(uint4*)&As[lrow * LDT + lks]     = a0;
            *(uint4*)&As[lrow * LDT + lks + 8] = a1;
            uint4 b0 = *(const uint4*)(sb);
            uint4 b1 = *(const uint4*)(sb + 8);
            *(uint4*)&Bs[lrow * LDT + lks]     = b0;
            *(uint4*)&Bs[lrow * LDT + lks + 8] = b1;
        }
#endif
        __syncthreads();

        v16bf af[4], bfrag[2];
#pragma unroll
        for (int tm = 0; tm < 4; ++tm)
            af[tm] = load_fragA(&As[(wm * 64 + tm * 16) * LDT], LDT, lane);
#pragma unroll
        for (int tn = 0; tn < 2; ++tn)
            bfrag[tn] = load_fragB(&Bs[(wn * 32 + tn * 16) * LDT], LDT, lane);
#pragma unroll
        for (int tm = 0; tm < 4; ++tm)
#pragma unroll
            for (int tn = 0; tn < 2; ++tn)
                acc[tm][tn] = __builtin_amdgcn_wmma_f32_16x16x32_bf16(
                    false, af[tm], false, bfrag[tn], (short)0, acc[tm][tn],
                    false, false);
    }

    // epilogue: acc VGPR j <-> row (j or j+8), col = lane&15
    const int rofs = (lane & 16) ? 8 : 0;
#pragma unroll
    for (int tm = 0; tm < 4; ++tm) {
#pragma unroll
        for (int tn = 0; tn < 2; ++tn) {
            int n = nBase + wn * 32 + tn * 16 + (lane & 15);
            float bn = bias[n];
#pragma unroll
            for (int j = 0; j < 8; ++j) {
                int m = mBase + wm * 64 + tm * 16 + rofs + j;
                float v = acc[tm][tn][j] + bn;
                if (mode == 0) {
                    int b = m >> 11, s = m & (SEQ - 1);
                    int h = n >> 6,  d = n & (DK - 1);
                    out_h[(((size_t)(b * NHEADS + h) * SEQ) + s) * DK + d] =
                        f32_to_bf16_bits(v);
                } else {
                    size_t idx = (size_t)m * N + n;
                    out_f32[idx] = v + residual[idx];
                }
            }
        }
    }
}

// ---------------------------------------------------------------------------
// Kernel: scores[b,h,i,j] = 0.125 * sum_d q[b,h,i,d]*k[b,h,j,d]  (+ mask)
// grid: (S/128, S/128, B*H). K=64 (two 32-steps).
// ---------------------------------------------------------------------------
__global__ void mha_scores(const unsigned short* __restrict__ qh,
                           const unsigned short* __restrict__ kh,
                           const unsigned char* __restrict__ mask,
                           float* __restrict__ scores) {
    __shared__ unsigned short As[128 * LDT];
    __shared__ unsigned short Bs[128 * LDT];

    const int t    = threadIdx.x;
    const int lane = t & 31;
    const int wave = t >> 5;
    const int wm   = wave >> 2;
    const int wn   = wave & 3;
    const int mBase = blockIdx.x * 128;
    const int nBase = blockIdx.y * 128;
    const int bh    = blockIdx.z;        // b*H + h

    const unsigned short* qb = qh + (size_t)bh * SEQ * DK;
    const unsigned short* kb = kh + (size_t)bh * SEQ * DK;

    v8f acc[4][2];
#pragma unroll
    for (int i = 0; i < 4; ++i)
#pragma unroll
        for (int j = 0; j < 2; ++j) acc[i][j] = (v8f){0,0,0,0,0,0,0,0};

    const int lrow = t >> 1;
    const int lks  = (t & 1) * 16;

    for (int k0 = 0; k0 < DK; k0 += 32) {
        const unsigned short* sa = qb + (size_t)(mBase + lrow) * DK + k0 + lks;
        const unsigned short* sb = kb + (size_t)(nBase + lrow) * DK + k0 + lks;
        __syncthreads();
#ifdef HAVE_ASYNC_LDS
        ASYNC_CP_B128(sa,     &As[lrow * LDT + lks]);
        ASYNC_CP_B128(sa + 8, &As[lrow * LDT + lks + 8]);
        ASYNC_CP_B128(sb,     &Bs[lrow * LDT + lks]);
        ASYNC_CP_B128(sb + 8, &Bs[lrow * LDT + lks + 8]);
        __builtin_amdgcn_s_wait_asynccnt(0);
#else
        {
            uint4 a0 = *(const uint4*)(sa);
            uint4 a1 = *(const uint4*)(sa + 8);
            *(uint4*)&As[lrow * LDT + lks]     = a0;
            *(uint4*)&As[lrow * LDT + lks + 8] = a1;
            uint4 b0 = *(const uint4*)(sb);
            uint4 b1 = *(const uint4*)(sb + 8);
            *(uint4*)&Bs[lrow * LDT + lks]     = b0;
            *(uint4*)&Bs[lrow * LDT + lks + 8] = b1;
        }
#endif
        __syncthreads();

        v16bf af[4], bfrag[2];
#pragma unroll
        for (int tm = 0; tm < 4; ++tm)
            af[tm] = load_fragA(&As[(wm * 64 + tm * 16) * LDT], LDT, lane);
#pragma unroll
        for (int tn = 0; tn < 2; ++tn)
            bfrag[tn] = load_fragB(&Bs[(wn * 32 + tn * 16) * LDT], LDT, lane);
#pragma unroll
        for (int tm = 0; tm < 4; ++tm)
#pragma unroll
            for (int tn = 0; tn < 2; ++tn)
                acc[tm][tn] = __builtin_amdgcn_wmma_f32_16x16x32_bf16(
                    false, af[tm], false, bfrag[tn], (short)0, acc[tm][tn],
                    false, false);
    }

    const int b    = bh >> 4;
    const int rofs = (lane & 16) ? 8 : 0;
    float* sout = scores + (size_t)bh * SEQ * SEQ;
    const unsigned char* mrow = mask + (size_t)b * SEQ * SEQ;
#pragma unroll
    for (int tm = 0; tm < 4; ++tm) {
#pragma unroll
        for (int tn = 0; tn < 2; ++tn) {
            int j = nBase + wn * 32 + tn * 16 + (lane & 15);
#pragma unroll
            for (int r = 0; r < 8; ++r) {
                int i = mBase + wm * 64 + tm * 16 + rofs + r;
                float v = 0.125f * acc[tm][tn][r];   // 1/sqrt(64)
                if (mrow[(size_t)i * SEQ + j]) v = NEGVAL;
                sout[(size_t)i * SEQ + j] = v;
            }
        }
    }
}

// ---------------------------------------------------------------------------
// Kernel: row softmax over 2048 cols. One block (256 thr) per row.
// ---------------------------------------------------------------------------
__global__ void mha_softmax(const float* __restrict__ scores,
                            float* __restrict__ attn) {
    __shared__ float red[256];
    const int t   = threadIdx.x;
    const size_t row = blockIdx.x;
    const float* src = scores + row * SEQ;
    float* dst       = attn   + row * SEQ;

    float x[8];
    float mx = -3.4e38f;
#pragma unroll
    for (int i = 0; i < 8; ++i) {
        x[i] = src[t * 8 + i];
        mx = fmaxf(mx, x[i]);
    }
    red[t] = mx;
    __syncthreads();
    for (int s = 128; s > 0; s >>= 1) {
        if (t < s) red[t] = fmaxf(red[t], red[t + s]);
        __syncthreads();
    }
    mx = red[0];
    __syncthreads();

    float sum = 0.0f;
#pragma unroll
    for (int i = 0; i < 8; ++i) {
        x[i] = __expf(x[i] - mx);
        sum += x[i];
    }
    red[t] = sum;
    __syncthreads();
    for (int s = 128; s > 0; s >>= 1) {
        if (t < s) red[t] += red[t + s];
        __syncthreads();
    }
    float inv = 1.0f / red[0];
#pragma unroll
    for (int i = 0; i < 8; ++i) dst[t * 8 + i] = x[i] * inv;
}

// ---------------------------------------------------------------------------
// Kernel: context[b,h,i,d] = sum_j attn[b,h,i,j] * v[b,h,j,d]
// M=2048, N=64, K=2048. grid (S/128, 1, B*H). A = fp32 attn (cvt->bf16 on
// stage), B = v (transposed through LDS to [d][j]).
// Output: bf16 model layout ctx[(b*S+i)*D + h*64 + d].
// ---------------------------------------------------------------------------
__global__ void mha_context(const float* __restrict__ attn,
                            const unsigned short* __restrict__ vh,
                            unsigned short* __restrict__ ctx) {
    __shared__ unsigned short As[128 * LDT];
    __shared__ unsigned short Bs[64 * LDT];

    const int t    = threadIdx.x;
    const int lane = t & 31;
    const int wave = t >> 5;
    const int wm   = wave >> 1;          // 0..3 -> 32 rows each
    const int wn   = wave & 1;           // 0..1 -> 32 cols each
    const int mBase = blockIdx.x * 128;
    const int bh    = blockIdx.z;

    const float* ab = attn + (size_t)bh * SEQ * SEQ;
    const unsigned short* vb = vh + (size_t)bh * SEQ * DK;

    v8f acc[2][2];
#pragma unroll
    for (int i = 0; i < 2; ++i)
#pragma unroll
        for (int j = 0; j < 2; ++j) acc[i][j] = (v8f){0,0,0,0,0,0,0,0};

    for (int k0 = 0; k0 < SEQ; k0 += 32) {
        __syncthreads();
        {   // A: 128x32 fp32 -> bf16. 2 threads/row, 16 floats each.
            int row = t >> 1, half = t & 1;
            const float* sa = ab + (size_t)(mBase + row) * SEQ + k0 + half * 16;
            unsigned short h[16];
#pragma unroll
            for (int q = 0; q < 4; ++q) {
                float4 f = *(const float4*)(sa + q * 4);
                h[q * 4 + 0] = f32_to_bf16_bits(f.x);
                h[q * 4 + 1] = f32_to_bf16_bits(f.y);
                h[q * 4 + 2] = f32_to_bf16_bits(f.z);
                h[q * 4 + 3] = f32_to_bf16_bits(f.w);
            }
            unsigned int* dst = (unsigned int*)&As[row * LDT + half * 16];
#pragma unroll
            for (int q = 0; q < 8; ++q)
                dst[q] = (unsigned int)h[2 * q] | ((unsigned int)h[2 * q + 1] << 16);

            // B: v tile 32(j) x 64(d) bf16 -> transpose into Bs[d][j]
            int jj = t >> 3, db = (t & 7) * 8;
            uint4 vv = *(const uint4*)(vb + ((size_t)(k0 + jj)) * DK + db);
            const unsigned short* vs = (const unsigned short*)&vv;
#pragma unroll
            for (int q = 0; q < 8; ++q)
                Bs[(db + q) * LDT + jj] = vs[q];
        }
        __syncthreads();

        v16bf af[2], bfrag[2];
#pragma unroll
        for (int tm = 0; tm < 2; ++tm)
            af[tm] = load_fragA(&As[(wm * 32 + tm * 16) * LDT], LDT, lane);
#pragma unroll
        for (int tn = 0; tn < 2; ++tn)
            bfrag[tn] = load_fragB(&Bs[(wn * 32 + tn * 16) * LDT], LDT, lane);
#pragma unroll
        for (int tm = 0; tm < 2; ++tm)
#pragma unroll
            for (int tn = 0; tn < 2; ++tn)
                acc[tm][tn] = __builtin_amdgcn_wmma_f32_16x16x32_bf16(
                    false, af[tm], false, bfrag[tn], (short)0, acc[tm][tn],
                    false, false);
    }

    const int b = bh >> 4, h = bh & 15;
    const int rofs = (lane & 16) ? 8 : 0;
#pragma unroll
    for (int tm = 0; tm < 2; ++tm) {
#pragma unroll
        for (int tn = 0; tn < 2; ++tn) {
            int d = wn * 32 + tn * 16 + (lane & 15);
#pragma unroll
            for (int r = 0; r < 8; ++r) {
                int i = mBase + wm * 32 + tm * 16 + rofs + r;
                ctx[((size_t)(b * SEQ + i)) * DMODEL + h * DK + d] =
                    f32_to_bf16_bits(acc[tm][tn][r]);
            }
        }
    }
}

// ---------------------------------------------------------------------------
// Launch
// ---------------------------------------------------------------------------
extern "C" void kernel_launch(void* const* d_in, const int* in_sizes, int n_in,
                              void* d_out, int out_size, void* d_ws, size_t ws_size,
                              hipStream_t stream) {
    const float* Qin = (const float*)d_in[0];
    const float* Kin = (const float*)d_in[1];
    const float* Vin = (const float*)d_in[2];
    const unsigned char* mask = (const unsigned char*)d_in[3];
    const float* Wq = (const float*)d_in[4];  const float* bq = (const float*)d_in[5];
    const float* Wk = (const float*)d_in[6];  const float* bk = (const float*)d_in[7];
    const float* Wv = (const float*)d_in[8];  const float* bv = (const float*)d_in[9];
    const float* Wo = (const float*)d_in[10]; const float* bo = (const float*)d_in[11];

    float* out  = (float*)d_out;                          // [B,S,D]
    float* attn = out + (size_t)MTOT * DMODEL;            // [B,H,S,S]
    float* scr  = attn + (size_t)B_SZ * NHEADS * SEQ * SEQ;

    // workspace carve-up (bf16 buffers)
    char* p = (char*)d_ws;
    size_t off = 0;
    auto carve = [&](size_t bytes) { char* r = p + off; off = (off + bytes + 255) & ~(size_t)255; return r; };
    const size_t xsz = (size_t)MTOT * DMODEL * 2;         // 8 MB
    const size_t wsz = (size_t)DMODEL * DMODEL * 2;       // 2 MB
    unsigned short* xq = (unsigned short*)carve(xsz);
    unsigned short* xk = (unsigned short*)carve(xsz);
    unsigned short* xv = (unsigned short*)carve(xsz);
    unsigned short* wqb = (unsigned short*)carve(wsz);
    unsigned short* wkb = (unsigned short*)carve(wsz);
    unsigned short* wvb = (unsigned short*)carve(wsz);
    unsigned short* wob = (unsigned short*)carve(wsz);
    unsigned short* qh  = (unsigned short*)carve(xsz);    // [B,H,S,DK]
    unsigned short* kh  = (unsigned short*)carve(xsz);
    unsigned short* vh  = (unsigned short*)carve(xsz);
    unsigned short* ctx = (unsigned short*)carve(xsz);    // [B*S, D]
    (void)ws_size; (void)n_in; (void)in_sizes; (void)out_size;

    const int nX = MTOT * DMODEL;       // 4194304
    const int nW = DMODEL * DMODEL;     // 1048576
    mha_cvt_bf16<<<(nX + 255) / 256, 256, 0, stream>>>(Qin, xq, nX);
    mha_cvt_bf16<<<(nX + 255) / 256, 256, 0, stream>>>(Kin, xk, nX);
    mha_cvt_bf16<<<(nX + 255) / 256, 256, 0, stream>>>(Vin, xv, nX);
    mha_cvt_bf16<<<(nW + 255) / 256, 256, 0, stream>>>(Wq, wqb, nW);
    mha_cvt_bf16<<<(nW + 255) / 256, 256, 0, stream>>>(Wk, wkb, nW);
    mha_cvt_bf16<<<(nW + 255) / 256, 256, 0, stream>>>(Wv, wvb, nW);
    mha_cvt_bf16<<<(nW + 255) / 256, 256, 0, stream>>>(Wo, wob, nW);

    dim3 gProj(MTOT / 128, DMODEL / 128);                 // 32 x 8
    mha_gemm_bf16<<<gProj, 256, 0, stream>>>(xq, wqb, bq, nullptr, nullptr, qh,
                                             MTOT, DMODEL, DMODEL, 0);
    mha_gemm_bf16<<<gProj, 256, 0, stream>>>(xk, wkb, bk, nullptr, nullptr, kh,
                                             MTOT, DMODEL, DMODEL, 0);
    mha_gemm_bf16<<<gProj, 256, 0, stream>>>(xv, wvb, bv, nullptr, nullptr, vh,
                                             MTOT, DMODEL, DMODEL, 0);

    dim3 gScore(SEQ / 128, SEQ / 128, B_SZ * NHEADS);     // 16 x 16 x 32
    mha_scores<<<gScore, 256, 0, stream>>>(qh, kh, mask, scr);

    mha_softmax<<<B_SZ * NHEADS * SEQ, 256, 0, stream>>>(scr, attn);

    dim3 gCtx(SEQ / 128, 1, B_SZ * NHEADS);               // 16 x 1 x 32
    mha_context<<<gCtx, 256, 0, stream>>>(attn, vh, ctx);

    mha_gemm_bf16<<<gProj, 256, 0, stream>>>(ctx, wob, bo, Qin, out, nullptr,
                                             MTOT, DMODEL, DMODEL, 1);
}
